// PolicyNetwork_50981261803601
// MI455X (gfx1250) — matrix operand order repaired and checked by reading, not compiled, not run on hardware
//
#include <hip/hip_runtime.h>
#include <math.h>

// ---------------- config ----------------
#define BB   512
#define NN   512
#define DD   128
#define HH   512
#define LL   8
#define MT   64            // tokens per workgroup tile
#define HSTRIDE 520        // bf16 elems per LDS row (512 + 8 pad -> 4-bank skew/row)
#define OSTR 132           // f32 out-row stride in LDS
#define INV_SQRT_D 0.08838834764831845f

typedef __attribute__((ext_vector_type(16))) __bf16 v16bf;
typedef __attribute__((ext_vector_type(2)))  __bf16 v2bf;
typedef __attribute__((ext_vector_type(8)))  float  v8f;

union ABf { v16bf v; uint4 q[2]; };

__device__ __forceinline__ unsigned short f2bf_sw(float f) {
    union { float f; unsigned u; } v; v.f = f;
    unsigned u = v.u;
    unsigned r = (u + 0x7FFFu + ((u >> 16) & 1u)) >> 16;   // RNE
    return (unsigned short)r;
}
// pack two f32 -> packed bf16 pair (single v_cvt_pk_bf16_f32 when available)
__device__ __forceinline__ unsigned pk2(float lo, float hi) {
#if __has_builtin(__builtin_amdgcn_cvt_pk_bf16_f32)
    union { v2bf v; unsigned u; } c;
    c.v = __builtin_amdgcn_cvt_pk_bf16_f32(lo, hi);
    return c.u;
#else
    return (unsigned)f2bf_sw(lo) | ((unsigned)f2bf_sw(hi) << 16);
#endif
}
// single f32 -> bf16 in the low 16 bits (b16 store consumes low half)
__device__ __forceinline__ unsigned short f2bf(float f) {
    return (unsigned short)(pk2(f, f) & 0xffffu);
}
// unpack a packed pair of bf16 (lo, hi) to floats
__device__ __forceinline__ void unpk2(unsigned u, float& lo, float& hi) {
    union { unsigned u; float f; } x, y;
    x.u = u << 16; y.u = u & 0xffff0000u;
    lo = x.f; hi = y.f;
}

// ---------- f32 -> bf16 weight conversion (4 elems/thread) ----------
__global__ void cvt_bf16_kernel(const float* __restrict__ src,
                                unsigned short* __restrict__ dst, int n4) {
    int i = blockIdx.x * 256 + threadIdx.x;
    if (i < n4) {
        float4 f = ((const float4*)src)[i];
        uint2 o;
        o.x = pk2(f.x, f.y);
        o.y = pk2(f.z, f.w);
        ((uint2*)dst)[i] = o;
    }
}

// ---------- wave-level GEMM: dest = relu(src @ W^T + bias) (or final f32 store) ----------
// Four independent accumulator chains: N-tile pair x K even/odd split. Dependent
// same-chain WMMAs are >=4 issue slots apart -> no hazard NOPs.
// B-fragment (32x16 bf16): lanes 0-15 hold k=0..15 of col n=lane; lanes 16-31 k=16..31.
// A-fragment (16x32 bf16): lanes 0-15 row m=lane, k=0..15; lanes 16-31 k=16..31.
template <bool FINAL>
__device__ __forceinline__ void wave_gemm(
    const unsigned short* src, unsigned short* sdst,
    float* fdst, int fstride,
    const unsigned short* __restrict__ Wrow, int Krow, int Kin,
    const float* __restrict__ bias, int Ntiles,
    float* rmean, float* rrstd, int wave, int lane)
{
    const int lcol  = lane & 15;
    const int lhalf = lane >> 4;
    const int arow  = wave * 16 + lcol;      // A row this lane supplies
    const int mbase = wave * 16 + lhalf * 8; // first C row this lane holds

    float sum[8], sq[8];
#pragma unroll
    for (int j = 0; j < 8; ++j) { sum[j] = 0.f; sq[j] = 0.f; }

    const unsigned short* ap0 = src + (size_t)arow * HSTRIDE + lhalf * 16;

    for (int nt = 0; nt < Ntiles; nt += 2) {
        const int n0 = nt * 16 + lcol;
        const int n1 = n0 + 16;
        const unsigned short* bp0 = Wrow + (size_t)n0 * Krow + lhalf * 16;
        const unsigned short* bp1 = Wrow + (size_t)n1 * Krow + lhalf * 16;

        v8f c0a = {}, c0b = {}, c1a = {}, c1b = {};
        for (int kk = 0; kk < Kin; kk += 64) {
            ABf a0, a1;
            a0.q[0] = *(const uint4*)(ap0 + kk);            // k 0..7   (or 16..23)
            a0.q[1] = *(const uint4*)(ap0 + kk + 8);        // k 8..15  (or 24..31)
            a1.q[0] = *(const uint4*)(ap0 + kk + 32);
            a1.q[1] = *(const uint4*)(ap0 + kk + 40);
            v16bf b0a = *(const v16bf*)(bp0 + kk);
            v16bf b1a = *(const v16bf*)(bp1 + kk);
            v16bf b0b = *(const v16bf*)(bp0 + kk + 32);
            v16bf b1b = *(const v16bf*)(bp1 + kk + 32);
            c0a = __builtin_amdgcn_wmma_f32_16x16x32_bf16(
                      false, a0.v, false, b0a, (short)0, c0a, false, false);
            c1a = __builtin_amdgcn_wmma_f32_16x16x32_bf16(
                      false, a0.v, false, b1a, (short)0, c1a, false, false);
            c0b = __builtin_amdgcn_wmma_f32_16x16x32_bf16(
                      false, a1.v, false, b0b, (short)0, c0b, false, false);
            c1b = __builtin_amdgcn_wmma_f32_16x16x32_bf16(
                      false, a1.v, false, b1b, (short)0, c1b, false, false);
        }
        v8f c0 = c0a + c0b;
        v8f c1 = c1a + c1b;

        float bn0 = bias[n0];
        float bn1 = bias[n1];
        if (FINAL) {
#pragma unroll
            for (int j = 0; j < 8; ++j) {
                fdst[(size_t)(mbase + j) * fstride + n0] = c0[j] + bn0;
                fdst[(size_t)(mbase + j) * fstride + n1] = c1[j] + bn1;
            }
        } else {
#pragma unroll
            for (int j = 0; j < 8; ++j) {
                float r0 = c0[j] + bn0; r0 = r0 > 0.f ? r0 : 0.f;
                float r1 = c1[j] + bn1; r1 = r1 > 0.f ? r1 : 0.f;
                sum[j] += r0 + r1; sq[j] += r0 * r0 + r1 * r1;
                unsigned short* dr = sdst + (size_t)(mbase + j) * HSTRIDE;
                dr[n0] = f2bf(r0);
                dr[n1] = f2bf(r1);
            }
        }
    }

    if (!FINAL) {
        const float Hn = (float)(Ntiles * 16);
#pragma unroll
        for (int j = 0; j < 8; ++j) {
            float s = sum[j], q = sq[j];
#pragma unroll
            for (int off = 8; off; off >>= 1) {        // reduce across the 16-lane group
                s += __shfl_xor(s, off, 16);
                q += __shfl_xor(q, off, 16);
            }
            if (lcol == 0) {
                float mu  = s / Hn;
                float var = q / Hn - mu * mu;
                rmean[mbase + j] = mu;
                rrstd[mbase + j] = rsqrtf(var + 1e-5f);
            }
        }
    }
}

// ---------- LayerNorm (+optional residual) applied in LDS, 8 elems/iter ----------
__device__ __forceinline__ void ln_apply(
    unsigned short* hb, const unsigned short* sb,
    const float* rmean, const float* rrstd,
    const float* __restrict__ g, const float* __restrict__ be,
    bool residual, int tid)
{
    int row = tid >> 1;
    int c0  = (tid & 1) * (HH / 2);
    float mu = rmean[row], rs = rrstd[row];
    unsigned short*       hr = hb + (size_t)row * HSTRIDE;
    const unsigned short* sr = sb + (size_t)row * HSTRIDE;

    for (int c = c0; c < c0 + HH / 2; c += 8) {
        uint4 sv = *(const uint4*)(sr + c);
        float s0, s1, s2, s3, s4, s5, s6, s7;
        unpk2(sv.x, s0, s1); unpk2(sv.y, s2, s3);
        unpk2(sv.z, s4, s5); unpk2(sv.w, s6, s7);
        float4 ga = *(const float4*)(g + c);
        float4 gb = *(const float4*)(g + c + 4);
        float4 ba = *(const float4*)(be + c);
        float4 bb = *(const float4*)(be + c + 4);
        float v0 = (s0 - mu) * rs * ga.x + ba.x;
        float v1 = (s1 - mu) * rs * ga.y + ba.y;
        float v2 = (s2 - mu) * rs * ga.z + ba.z;
        float v3 = (s3 - mu) * rs * ga.w + ba.w;
        float v4 = (s4 - mu) * rs * gb.x + bb.x;
        float v5 = (s5 - mu) * rs * gb.y + bb.y;
        float v6 = (s6 - mu) * rs * gb.z + bb.z;
        float v7 = (s7 - mu) * rs * gb.w + bb.w;
        if (residual) {
            uint4 hv = *(const uint4*)(hr + c);
            float h0, h1, h2, h3, h4, h5, h6, h7;
            unpk2(hv.x, h0, h1); unpk2(hv.y, h2, h3);
            unpk2(hv.z, h4, h5); unpk2(hv.w, h6, h7);
            v0 += h0; v1 += h1; v2 += h2; v3 += h3;
            v4 += h4; v5 += h5; v6 += h6; v7 += h7;
        }
        uint4 ov;
        ov.x = pk2(v0, v1); ov.y = pk2(v2, v3);
        ov.z = pk2(v4, v5); ov.w = pk2(v6, v7);
        *(uint4*)(hr + c) = ov;
    }
}

// ---------- fused encoder (MODE 0: obs -> enc_obs ws; MODE 1: actions -> logits) ----------
template <int MODE>
__global__ __launch_bounds__(128, 1) void encode_kernel(
    const float* __restrict__ x_in,
    const unsigned char* __restrict__ mask,
    const unsigned short* __restrict__ Wb,
    const float* __restrict__ b0,   const float* __restrict__ g0,   const float* __restrict__ be0,
    const float* __restrict__ bres, const float* __restrict__ gres, const float* __restrict__ beres,
    const float* __restrict__ b1,   const float* __restrict__ b2,
    float* __restrict__ enc_obs, float* __restrict__ out)
{
    __shared__ __align__(32) unsigned short hbuf[MT * HSTRIDE];
    __shared__ __align__(32) unsigned short sbuf[MT * HSTRIDE];
    __shared__ float rmean[MT], rrstd[MT];
    __shared__ float eobs[DD];

    const unsigned short* W0b = Wb;
    const unsigned short* Wrb = W0b + (size_t)HH * DD;            // 65536
    const unsigned short* W1b = Wrb + (size_t)LL * HH * HH;       // +2097152
    const unsigned short* W2b = W1b + (size_t)HH * HH;            // +262144

    const int t0   = blockIdx.x * MT;
    const int tid  = threadIdx.x;
    const int wave = tid >> 5;
    const int lane = tid & 31;

    // load 64xD input tile as bf16 into hbuf (4 f32 -> packed pair of pairs)
    for (int i = tid; i < MT * (DD / 4); i += 128) {
        int m = i / (DD / 4), kq = i % (DD / 4);
        float4 f = ((const float4*)(x_in + (size_t)(t0 + m) * DD))[kq];
        uint2 o; o.x = pk2(f.x, f.y); o.y = pk2(f.z, f.w);
        *(uint2*)(hbuf + (size_t)m * HSTRIDE + kq * 4) = o;
    }
    if (MODE == 1) {
        int bidx = t0 / NN;                                       // whole tile shares b
        if (tid < DD) eobs[tid] = enc_obs[(size_t)bidx * DD + tid];
    }
    __syncthreads();

    // layer 0: h = LN(relu(x @ W0^T + b0))
    wave_gemm<false>(hbuf, sbuf, nullptr, 0, W0b, DD, DD, b0, HH / 16,
                     rmean, rrstd, wave, lane);
    __syncthreads();
    ln_apply(hbuf, sbuf, rmean, rrstd, g0, be0, false, tid);
    __syncthreads();

    // 8 residual blocks: h = h + LN(relu(h @ W^T + b))
    for (int l = 0; l < LL; ++l) {
        wave_gemm<false>(hbuf, sbuf, nullptr, 0, Wrb + (size_t)l * HH * HH, HH, HH,
                         bres + l * HH, HH / 16, rmean, rrstd, wave, lane);
        __syncthreads();
        ln_apply(hbuf, sbuf, rmean, rrstd, gres + l * HH, beres + l * HH, true, tid);
        __syncthreads();
    }

    // head: o1 = relu(h @ W1^T + b1) -> sbuf
    wave_gemm<false>(hbuf, sbuf, nullptr, 0, W1b, HH, HH, b1, HH / 16,
                     rmean, rrstd, wave, lane);
    __syncthreads();

    // out = o1 @ W2^T + b2 -> obuf (f32, aliases hbuf)
    float* obuf = (float*)hbuf;
    wave_gemm<true>(sbuf, nullptr, obuf, OSTR, W2b, HH, HH, b2, DD / 16,
                    rmean, rrstd, wave, lane);
    __syncthreads();

    if (MODE == 0) {
        for (int i = tid; i < MT * (DD / 4); i += 128) {
            int m = i / (DD / 4), dq = i % (DD / 4);
            ((float4*)(enc_obs + (size_t)(t0 + m) * DD))[dq] =
                ((const float4*)(obuf + (size_t)m * OSTR))[dq];
        }
    } else {
        if (tid < MT) {
            int t = t0 + tid;
            const float4* orow = (const float4*)(obuf + (size_t)tid * OSTR);
            const float4* ev   = (const float4*)eobs;
            float acc = 0.f;
#pragma unroll 8
            for (int d = 0; d < DD / 4; ++d) {
                float4 o = orow[d], e = ev[d];
                acc += o.x * e.x + o.y * e.y + o.z * e.z + o.w * e.w;
            }
            out[t] = (mask[t] != 0) ? acc * INV_SQRT_D : -__builtin_inff();
        }
    }
}

// ---------------- host launcher ----------------
extern "C" void kernel_launch(void* const* d_in, const int* in_sizes, int n_in,
                              void* d_out, int out_size, void* d_ws, size_t ws_size,
                              hipStream_t stream) {
    const float* obs   = (const float*)d_in[0];
    const float* act   = (const float*)d_in[1];
    const unsigned char* mask = (const unsigned char*)d_in[2];
    const float* W0    = (const float*)d_in[3];
    const float* b0    = (const float*)d_in[4];
    const float* g0    = (const float*)d_in[5];
    const float* be0   = (const float*)d_in[6];
    const float* Wres  = (const float*)d_in[7];
    const float* bres  = (const float*)d_in[8];
    const float* gres  = (const float*)d_in[9];
    const float* beres = (const float*)d_in[10];
    const float* W1    = (const float*)d_in[11];
    const float* b1    = (const float*)d_in[12];
    const float* W2    = (const float*)d_in[13];
    const float* b2    = (const float*)d_in[14];

    const size_t nW0 = (size_t)HH * DD;          // 65536
    const size_t nWr = (size_t)LL * HH * HH;     // 2097152
    const size_t nW1 = (size_t)HH * HH;          // 262144
    const size_t nW2 = (size_t)DD * HH;          // 65536
    const size_t nWtot = nW0 + nWr + nW1 + nW2;  // 2490368

    unsigned short* Wb = (unsigned short*)d_ws;
    float* enc_obs = (float*)((char*)d_ws + nWtot * sizeof(unsigned short)); // 4,980,736 B (256-aligned)

    // f32 -> bf16 weight conversion (4 elems per thread)
    cvt_bf16_kernel<<<(int)((nW0 / 4 + 255) / 256), 256, 0, stream>>>(W0,   Wb,                   (int)(nW0 / 4));
    cvt_bf16_kernel<<<(int)((nWr / 4 + 255) / 256), 256, 0, stream>>>(Wres, Wb + nW0,             (int)(nWr / 4));
    cvt_bf16_kernel<<<(int)((nW1 / 4 + 255) / 256), 256, 0, stream>>>(W1,   Wb + nW0 + nWr,       (int)(nW1 / 4));
    cvt_bf16_kernel<<<(int)((nW2 / 4 + 255) / 256), 256, 0, stream>>>(W2,   Wb + nW0 + nWr + nW1, (int)(nW2 / 4));

    // encode obs (512 tokens -> enc_obs ws)
    encode_kernel<0><<<BB / MT, 128, 0, stream>>>(
        obs, nullptr, Wb, b0, g0, be0, bres, gres, beres, b1, b2,
        enc_obs, nullptr);

    // encode actions + fused masked logits
    encode_kernel<1><<<(BB * NN) / MT, 128, 0, stream>>>(
        act, mask, Wb, b0, g0, be0, bres, gres, beres, b1, b2,
        enc_obs, (float*)d_out);
}
